// RGCN_10393820857054
// MI455X (gfx1250) — matrix operand order repaired and checked by reading, not compiled, not run on hardware
//
#include <hip/hip_runtime.h>

#define NN       50000
#define EDIM     128
#define PDIM     768
#define HID      256
#define NREL     16
#define NEDGE    400000
#define BN_EPS   1e-5f

typedef __bf16 bf16_t;
typedef __attribute__((ext_vector_type(16))) __bf16 v16bf;
typedef __attribute__((ext_vector_type(8)))  __bf16 v8bf;
typedef __attribute__((ext_vector_type(8)))  float  v8f;

// ---------------------------------------------------------------------------
// 1) Row-wise L2 normalize emb -> bf16 (stride EDIM)
// ---------------------------------------------------------------------------
__global__ void __launch_bounds__(128)
k_normalize_bf16(const float* __restrict__ emb, bf16_t* __restrict__ out)
{
    __shared__ float red[128];
    const int n = blockIdx.x, t = threadIdx.x;
    const float v = emb[(size_t)n * EDIM + t];
    red[t] = v * v;
    __syncthreads();
    #pragma unroll
    for (int s = 64; s > 0; s >>= 1) {
        if (t < s) red[t] += red[t + s];
        __syncthreads();
    }
    const float inv = 1.0f / fmaxf(sqrtf(red[0]), 1e-12f);
    out[(size_t)n * EDIM + t] = (bf16_t)(v * inv);
}

// ---------------------------------------------------------------------------
// 2) Weight prep: f32 [K x Nc] row-major -> bf16 transposed [Nc x K]
// ---------------------------------------------------------------------------
__global__ void __launch_bounds__(256)
k_transpose_bf16(const float* __restrict__ W, bf16_t* __restrict__ BT, int K, int Nc)
{
    const size_t total = (size_t)K * Nc;
    for (size_t i = (size_t)blockIdx.x * blockDim.x + threadIdx.x; i < total;
         i += (size_t)gridDim.x * blockDim.x) {
        const int k = (int)(i / Nc);
        const int c = (int)(i % Nc);
        BT[(size_t)c * K + k] = (bf16_t)W[i];
    }
}

// Pack [bases0 | bases1 | root] -> BT bf16 [768 x d_in]
__global__ void __launch_bounds__(256)
k_pack_layer_w(const float* __restrict__ bases, const float* __restrict__ root,
               bf16_t* __restrict__ BT, int d_in)
{
    const size_t total = (size_t)768 * d_in;
    for (size_t i = (size_t)blockIdx.x * blockDim.x + threadIdx.x; i < total;
         i += (size_t)gridDim.x * blockDim.x) {
        const int c = (int)(i / d_in);
        const int k = (int)(i % d_in);
        float s;
        if (c < HID)            s = bases[(size_t)k * HID + c];
        else if (c < 2 * HID)   s = bases[(size_t)d_in * HID + (size_t)k * HID + (c - HID)];
        else                    s = root[(size_t)k * HID + (c - 2 * HID)];
        BT[(size_t)c * d_in + k] = (bf16_t)s;
    }
}

// ---------------------------------------------------------------------------
// 3) WMMA GEMM: C[M x Nc] (f32) = A[M x K] (bf16, row-major)
//                               x BT[Nc x K] (bf16, K-contiguous) (+ bias)
// Register-blocked: one wave computes a 16x64 strip (4 accumulators), so each
// A fragment feeds 4 WMMAs.  Block = 4 waves = 64x64 C tile.
// Nc % 64 == 0, K % 32 == 0 at all call sites; row guard is wave-uniform so
// EXEC is all-ones around every WMMA.
// ---------------------------------------------------------------------------
__device__ __forceinline__ void store_tile(float* __restrict__ C, const v8f acc,
                                           int row0, int col, int Nc, int hi, int r,
                                           const float* __restrict__ bias)
{
    const float badd = bias ? bias[col + r] : 0.0f;
    float* __restrict__ crow = C + (size_t)(row0 + hi * 8) * Nc + col + r;
    #pragma unroll
    for (int v = 0; v < 8; ++v)          // C/D layout: VGPR v -> row (hi*8 + v)
        crow[(size_t)v * Nc] = acc[v] + badd;
}

__global__ void __launch_bounds__(128)
k_wmma_gemm_bf16(const bf16_t* __restrict__ A, const bf16_t* __restrict__ BT,
                 const float* __restrict__ bias, float* __restrict__ C,
                 int Nc, int K)
{
    const int wave = threadIdx.x >> 5;
    const int lane = threadIdx.x & 31;
    const int r    = lane & 15;          // output column lane
    const int hi   = lane >> 4;          // half-wave select
    const int row0 = (blockIdx.y * 4 + wave) << 4;
    if (row0 >= NN) return;              // wave-uniform: EXEC stays full
    const int col0 = blockIdx.x << 6;    // 64-column strip

    const bf16_t* __restrict__ arow = A  + (size_t)(row0 + r) * K;
    const bf16_t* __restrict__ bcol = BT + (size_t)(col0 + r) * K;

    v8f acc0 = {}, acc1 = {}, acc2 = {}, acc3 = {};
    for (int k = 0; k < K; k += 32) {
        // A, contiguous k order used here: lanes 0-15 hold K = k..k+15 (v0-7);
        // lanes 16-31 hold K = k+16..k+31 (matches B's contiguous order).
        union { v16bf v; v8bf h[2]; } af;
        af.h[0] = *(const v8bf*)(arow + k + hi * 16);
        af.h[1] = *(const v8bf*)(arow + k + hi * 16 + 8);
        // 4 B fragments: lane's 16 consecutive K values of 4 output columns
        // (weights pre-transposed -> one contiguous 32B load each).
        const v16bf bv0 = *(const v16bf*)(bcol + k + hi * 16);
        const v16bf bv1 = *(const v16bf*)(bcol + (size_t)16 * K + k + hi * 16);
        const v16bf bv2 = *(const v16bf*)(bcol + (size_t)32 * K + k + hi * 16);
        const v16bf bv3 = *(const v16bf*)(bcol + (size_t)48 * K + k + hi * 16);
        acc0 = __builtin_amdgcn_wmma_f32_16x16x32_bf16(false, af.v, false, bv0, (short)0, acc0, false, false);
        acc1 = __builtin_amdgcn_wmma_f32_16x16x32_bf16(false, af.v, false, bv1, (short)0, acc1, false, false);
        acc2 = __builtin_amdgcn_wmma_f32_16x16x32_bf16(false, af.v, false, bv2, (short)0, acc2, false, false);
        acc3 = __builtin_amdgcn_wmma_f32_16x16x32_bf16(false, af.v, false, bv3, (short)0, acc3, false, false);
    }

    store_tile(C, acc0, row0, col0 +  0, Nc, hi, r, bias);
    store_tile(C, acc1, row0, col0 + 16, Nc, hi, r, bias);
    store_tile(C, acc2, row0, col0 + 32, Nc, hi, r, bias);
    store_tile(C, acc3, row0, col0 + 48, Nc, hi, r, bias);
}

// ---------------------------------------------------------------------------
// 4) Misc streaming kernels
// ---------------------------------------------------------------------------
__global__ void __launch_bounds__(256)
k_f32_to_bf16(const float* __restrict__ x, bf16_t* __restrict__ y, size_t n)
{
    for (size_t i = (size_t)blockIdx.x * blockDim.x + threadIdx.x; i < n;
         i += (size_t)gridDim.x * blockDim.x)
        y[i] = (bf16_t)x[i];
}

__global__ void __launch_bounds__(256)
k_fill_zero(float* __restrict__ p, size_t n)
{
    for (size_t i = (size_t)blockIdx.x * blockDim.x + threadIdx.x; i < n;
         i += (size_t)gridDim.x * blockDim.x)
        p[i] = 0.0f;
}

__global__ void __launch_bounds__(256)
k_count_edges(const int* __restrict__ dst, const int* __restrict__ et,
              float* __restrict__ cnt)
{
    for (int e = blockIdx.x * blockDim.x + threadIdx.x; e < NEDGE;
         e += gridDim.x * blockDim.x)
        atomicAdd(&cnt[(size_t)dst[e] * NREL + et[e]], 1.0f);
}

// out = H[:, 512:768] (x @ root) + bias
__global__ void __launch_bounds__(256)
k_init_out(const float* __restrict__ H, const float* __restrict__ bias,
           float* __restrict__ out)
{
    const size_t total = (size_t)NN * HID;
    for (size_t i = (size_t)blockIdx.x * blockDim.x + threadIdx.x; i < total;
         i += (size_t)gridDim.x * blockDim.x) {
        const size_t n = i >> 8;
        const int    c = (int)(i & (HID - 1));
        out[i] = H[n * PDIM + 2 * HID + c] + bias[c];
    }
}

// 8 edges per block, thread = output channel.  H (153.6MB) is L2-resident;
// global_prefetch_b8 pulls the NEXT edge's gather rows while the current
// edge's FMA + atomic is in flight.
#define EPB 8
__global__ void __launch_bounds__(256)
k_edge_scatter(const int* __restrict__ src, const int* __restrict__ dst,
               const int* __restrict__ et, const float* __restrict__ comp,
               const float* __restrict__ cnt, const float* __restrict__ H,
               float* __restrict__ out)
{
    const int t  = threadIdx.x;
    const int e0 = blockIdx.x * EPB;
    if (e0 >= NEDGE) return;

    int s_cur = src[e0];
    #pragma unroll
    for (int i = 0; i < EPB; ++i) {
        const int e = e0 + i;
        if (e >= NEDGE) break;
        const int s = s_cur;
        if (i + 1 < EPB && e + 1 < NEDGE) {
            s_cur = src[e + 1];
            __builtin_prefetch(H + (size_t)s_cur * PDIM + t, 0, 1);
            __builtin_prefetch(H + (size_t)s_cur * PDIM + HID + t, 0, 1);
        }
        const int d   = dst[e];
        const int rel = et[e];
        const float c0 = comp[rel * 2 + 0];
        const float c1 = comp[rel * 2 + 1];
        const float inv = 1.0f / fmaxf(cnt[(size_t)d * NREL + rel], 1.0f);
        const float m = (c0 * H[(size_t)s * PDIM + t] +
                         c1 * H[(size_t)s * PDIM + HID + t]) * inv;
        atomicAdd(&out[(size_t)d * HID + t], m);
    }
}

// Per-column sum / sumsq: thread = column, block = 128-row slab (coalesced).
__global__ void __launch_bounds__(256)
k_bn_stats(const float* __restrict__ x, float* __restrict__ stats)
{
    const int t = threadIdx.x;
    const int row0 = blockIdx.x * 128;
    float s = 0.0f, s2 = 0.0f;
    for (int i = 0; i < 128; ++i) {
        const int row = row0 + i;
        if (row < NN) {
            const float v = x[(size_t)row * HID + t];
            s += v; s2 += v * v;
        }
    }
    atomicAdd(&stats[t], s);
    atomicAdd(&stats[HID + t], s2);
}

// y = relu(BN(x)); writes f32 (final layer) and/or bf16 (next layer input)
__global__ void __launch_bounds__(256)
k_bn_apply(const float* __restrict__ x, const float* __restrict__ stats,
           const float* __restrict__ gamma, const float* __restrict__ beta,
           float* __restrict__ out_f32, bf16_t* __restrict__ out_bf16)
{
    const size_t total = (size_t)NN * HID;
    for (size_t i = (size_t)blockIdx.x * blockDim.x + threadIdx.x; i < total;
         i += (size_t)gridDim.x * blockDim.x) {
        const int c = (int)(i & (HID - 1));
        const float mu  = stats[c] * (1.0f / NN);
        const float var = stats[HID + c] * (1.0f / NN) - mu * mu;
        float y = (x[i] - mu) * rsqrtf(var + BN_EPS) * gamma[c] + beta[c];
        y = fmaxf(y, 0.0f);
        if (out_f32)  out_f32[i]  = y;
        if (out_bf16) out_bf16[i] = (bf16_t)y;
    }
}

// ---------------------------------------------------------------------------
// Host orchestration
// ---------------------------------------------------------------------------
extern "C" void kernel_launch(void* const* d_in, const int* in_sizes, int n_in,
                              void* d_out, int out_size, void* d_ws, size_t ws_size,
                              hipStream_t stream)
{
    (void)in_sizes; (void)n_in; (void)out_size; (void)ws_size;

    const int*   src    = (const int*)d_in[0];              // edge_index[0]
    const int*   dst    = (const int*)d_in[0] + NEDGE;      // edge_index[1]
    const int*   etype  = (const int*)d_in[1];
    const float* emb    = (const float*)d_in[2];
    const float* proj_w = (const float*)d_in[3];
    const float* proj_b = (const float*)d_in[4];

    const float* comp[3], *bases[3], *root[3], *bias[3], *gamma[3], *beta[3];
    for (int l = 0; l < 3; ++l) {
        const int b = 5 + l * 6;
        comp[l]  = (const float*)d_in[b + 0];
        bases[l] = (const float*)d_in[b + 1];
        root[l]  = (const float*)d_in[b + 2];
        bias[l]  = (const float*)d_in[b + 3];
        gamma[l] = (const float*)d_in[b + 4];
        beta[l]  = (const float*)d_in[b + 5];
    }

    // Workspace partition (~286 MB total)
    char* w = (char*)d_ws;
    bf16_t* XBF  = (bf16_t*)w;  w += (size_t)NN * PDIM * sizeof(bf16_t);   // 76.8 MB
    bf16_t* BT   = (bf16_t*)w;  w += (size_t)PDIM * PDIM * sizeof(bf16_t); // 1.2 MB
    float*  H    = (float*)w;   w += (size_t)NN * PDIM * sizeof(float);    // 153.6 MB (fits L2)
    float*  OUTB = (float*)w;   w += (size_t)NN * HID * sizeof(float);     // 51.2 MB
    float*  CNT  = (float*)w;   w += (size_t)NN * NREL * sizeof(float);    // 3.2 MB
    float*  STATS= (float*)w;   w += 2 * HID * sizeof(float);

    const int d_in_l[3] = { PDIM, HID, HID };
    float* outf = (float*)d_out;

    const int row_tiles  = NN / 16;                       // 3125
    const dim3 gemm_grid(PDIM / 64, (row_tiles + 3) / 4); // 12 x 782

    // --- input normalize + projection ---
    k_normalize_bf16<<<NN, 128, 0, stream>>>(emb, XBF);
    k_transpose_bf16<<<(EDIM * PDIM + 255) / 256, 256, 0, stream>>>(proj_w, BT, EDIM, PDIM);
    k_wmma_gemm_bf16<<<gemm_grid, 128, 0, stream>>>(XBF, BT, proj_b, H, PDIM, EDIM);
    k_f32_to_bf16<<<2048, 256, 0, stream>>>(H, XBF, (size_t)NN * PDIM);

    // --- per-(dst,rel) edge counts: edge-only, computed once ---
    k_fill_zero<<<1024, 256, 0, stream>>>(CNT, (size_t)NN * NREL);
    k_count_edges<<<(NEDGE + 255) / 256, 256, 0, stream>>>(dst, etype, CNT);

    // --- 3 RGCN layers ---
    for (int l = 0; l < 3; ++l) {
        const int K = d_in_l[l];
        k_pack_layer_w<<<(768 * K + 255) / 256, 256, 0, stream>>>(bases[l], root[l], BT, K);
        // H[:, 0:256]=x@B0, [:,256:512]=x@B1, [:,512:768]=x@root
        k_wmma_gemm_bf16<<<gemm_grid, 128, 0, stream>>>(XBF, BT, nullptr, H, PDIM, K);
        k_init_out<<<2048, 256, 0, stream>>>(H, bias[l], OUTB);
        k_edge_scatter<<<(NEDGE + EPB - 1) / EPB, 256, 0, stream>>>(src, dst, etype, comp[l], CNT, H, OUTB);
        k_fill_zero<<<2, 256, 0, stream>>>(STATS, 2 * HID);
        k_bn_stats<<<(NN + 127) / 128, 256, 0, stream>>>(OUTB, STATS);
        k_bn_apply<<<2048, 256, 0, stream>>>(OUTB, STATS, gamma[l], beta[l],
                                             (l == 2) ? outf : nullptr,
                                             (l < 2) ? XBF : nullptr);
    }
}
